// BahdanauAttention_38293928411174
// MI455X (gfx1250) — matrix operand order repaired and checked by reading, not compiled, not run on hardware
//
#include <hip/hip_runtime.h>

// ---------------- problem constants ----------------
constexpr int kB = 32;
constexpr int kS = 2048;
constexpr int kH = 1024;

constexpr int kSTileRows = 128;    // s-rows staged per block (4 waves x 2 tiles x 16)
constexpr int kWaves     = 4;      // wave32 waves per block; each owns 32 s-rows
constexpr int kAStride   = 1032;   // padded ushort stride (2064B) -> rows shift 4 banks
constexpr int kSChunk    = 8;      // s-splits for the context pass

typedef __attribute__((ext_vector_type(16))) __bf16 v16bf;
typedef __attribute__((ext_vector_type(8)))  float  v8f;

union BFrag {
    v16bf v;
    unsigned short u[16];
    uint4 q[2];
};

// round-to-nearest-even f32 -> bf16 (bit pattern)
__device__ __forceinline__ unsigned int f2bf(float x) {
    unsigned u = __float_as_uint(x);
    u += 0x7FFFu + ((u >> 16) & 1u);
    return u >> 16;
}

// branch-free tanh: clamp to +/-9 (tanh(9)==1.0 in fp32), then (e^2x-1)/(e^2x+1)
// lowers to v_max/v_min + v_exp_f32 + v_rcp_f32 straight-line code (no EXEC games)
__device__ __forceinline__ float fast_tanh(float x) {
    const float cx = fminf(fmaxf(x, -9.0f), 9.0f);
    const float e  = __expf(2.0f * cx);
    return (e - 1.0f) * __builtin_amdgcn_rcpf(e + 1.0f);
}

// ---------------------------------------------------------------------------
// Kernel 0: one-shot U_a f32 -> bf16 (removes ALL in-loop B-side conversion)
// ---------------------------------------------------------------------------
__global__ void cvt_ua_kernel(const float* __restrict__ U_a,
                              unsigned short* __restrict__ U_bf) {
    const int i = blockIdx.x * blockDim.x + threadIdx.x;   // kH*kH/4 threads
    const float4 v = ((const float4*)U_a)[i];
    uint2 p;
    p.x = f2bf(v.x) | (f2bf(v.y) << 16);
    p.y = f2bf(v.z) | (f2bf(v.w) << 16);
    ((uint2*)U_bf)[i] = p;
}

// ---------------------------------------------------------------------------
// Kernel 1: wh[b][o] = sum_h hidden[b][h] * W_a[o][h]        (tiny GEMV)
// ---------------------------------------------------------------------------
__global__ void wh_kernel(const float* __restrict__ hidden,
                          const float* __restrict__ W_a,
                          float* __restrict__ wh) {
    int idx = blockIdx.x * blockDim.x + threadIdx.x;   // kB*kH threads
    int b = idx >> 10;
    int o = idx & (kH - 1);
    const float4* hp = (const float4*)(hidden + (size_t)b * kH);
    const float4* wp = (const float4*)(W_a + (size_t)o * kH);
    float acc = 0.f;
    #pragma unroll 4
    for (int i = 0; i < kH / 4; ++i) {
        float4 hv = hp[i], wv = wp[i];
        acc += hv.x * wv.x + hv.y * wv.y + hv.z * wv.z + hv.w * wv.w;
    }
    wh[idx] = acc;
}

// ---------------------------------------------------------------------------
// Kernel 2 (WMMA, fused): score[b][s] = sum_o v_a[o]*tanh(wh[b][o] + ue[b][s][o])
// ue tiles live only in WMMA accumulators. Each wave owns TWO 16-row s-tiles
// that share one B fragment (2x WMMA per B load, independent back-to-back WMMAs).
// ---------------------------------------------------------------------------
__global__ void score_kernel(const float* __restrict__ enc,            // [B,S,H] f32
                             const unsigned short* __restrict__ U_bf,  // [H,H] bf16
                             const float* __restrict__ wh,             // [B,H]
                             const float* __restrict__ v_a,            // [H]
                             float* __restrict__ score)                // [B,S]
{
    extern __shared__ unsigned short A_lds[];   // [kSTileRows][kAStride]
    const int b   = blockIdx.y;
    const int s0  = blockIdx.x * kSTileRows;
    const int tid = threadIdx.x;

    // ---- stage A tile (128 x 1024 f32 -> bf16 in LDS), coalesced, packed stores
    const float* encBase = enc + ((size_t)b * kS + s0) * kH;
    for (int i = tid * 4; i < kSTileRows * kH; i += kWaves * 32 * 4) {
        const int row = i >> 10;
        const int col = i & (kH - 1);
        const float4 v = *(const float4*)(encBase + (size_t)row * kH + col);
        uint2 p;
        p.x = f2bf(v.x) | (f2bf(v.y) << 16);
        p.y = f2bf(v.z) | (f2bf(v.w) << 16);
        *(uint2*)(&A_lds[row * kAStride + col]) = p;
    }
    __syncthreads();

    const int lane = tid & 31;
    const int wave = tid >> 5;
    const int n    = lane & 15;        // WMMA column index (o within tile)
    const int hi   = lane >> 4;        // 0: lanes 0-15, 1: lanes 16-31

    // ISA 16-bit A layout: lane holds row M = lane&15;
    // elems 0..7 = K k0+hi*8 .. +7, elems 8..15 = K k0+16+hi*8 .. +7
    const unsigned short* Arow0 = &A_lds[(wave * 32 + n) * kAStride];
    const unsigned short* Arow1 = Arow0 + 16 * kAStride;

    float sacc0[8] = {0.f, 0.f, 0.f, 0.f, 0.f, 0.f, 0.f, 0.f};
    float sacc1[8] = {0.f, 0.f, 0.f, 0.f, 0.f, 0.f, 0.f, 0.f};

    for (int o0 = 0; o0 < kH; o0 += 16) {
        v8f c0 = {};
        v8f c1 = {};
        // B[k][n] = U_bf[o0+n][k0+k]; lane (N=n, half hi) holds K = k0+hi*8 .. +7, k0+16+hi*8 .. +7
        const unsigned short* Brow = U_bf + (size_t)(o0 + n) * kH + hi * 8;

        #pragma unroll 4
        for (int k0 = 0; k0 < kH; k0 += 32) {
            BFrag a0, a1, bm;
            bm.q[0] = *(const uint4*)(Brow + k0);        // 16B: K hi*8 .. +7
            bm.q[1] = *(const uint4*)(Brow + k0 + 16);   // 16B: K 16+hi*8 .. +7
            a0.q[0] = *(const uint4*)(Arow0 + k0 + hi * 8);
            a0.q[1] = *(const uint4*)(Arow0 + k0 + 16 + hi * 8);
            a1.q[0] = *(const uint4*)(Arow1 + k0 + hi * 8);
            a1.q[1] = *(const uint4*)(Arow1 + k0 + 16 + hi * 8);

            // (neg_a, A, neg_b, B, c_mod, C, reuse_a, reuse_b)
            c0 = __builtin_amdgcn_wmma_f32_16x16x32_bf16(
                     false, a0.v, false, bm.v, (short)0, c0, false, false);
            c1 = __builtin_amdgcn_wmma_f32_16x16x32_bf16(
                     false, a1.v, false, bm.v, (short)0, c1, false, false);
        }

        // epilogue in registers: + wh, tanh, * v_a, accumulate over o
        const float whv = wh[b * kH + o0 + n];
        const float vv  = v_a[o0 + n];
        #pragma unroll
        for (int r = 0; r < 8; ++r) {
            sacc0[r] += vv * fast_tanh(c0[r] + whv);
            sacc1[r] += vv * fast_tanh(c1[r] + whv);
        }
    }

    // C layout: VGPR r -> row M=r (lanes 0-15) / M=r+8 (lanes 16-31), N=lane&15.
    // Reduce over N: xor-shuffle within each 16-lane half.
    #pragma unroll
    for (int r = 0; r < 8; ++r) {
        float x = sacc0[r], y = sacc1[r];
        x += __shfl_xor(x, 1, 32);  y += __shfl_xor(y, 1, 32);
        x += __shfl_xor(x, 2, 32);  y += __shfl_xor(y, 2, 32);
        x += __shfl_xor(x, 4, 32);  y += __shfl_xor(y, 4, 32);
        x += __shfl_xor(x, 8, 32);  y += __shfl_xor(y, 8, 32);
        sacc0[r] = x; sacc1[r] = y;
    }
    if (n == 0) {   // lane 0 writes rows 0..7, lane 16 writes rows 8..15
        float* out = score + (size_t)b * kS + s0 + wave * 32 + hi * 8;
        #pragma unroll
        for (int r = 0; r < 8; ++r) {
            out[r]      = sacc0[r];
            out[r + 16] = sacc1[r];
        }
    }
}

// ---------------------------------------------------------------------------
// Kernel 3: softmax over S per batch row
// ---------------------------------------------------------------------------
__global__ void softmax_kernel(const float* __restrict__ score,
                               float* __restrict__ attn) {
    __shared__ float red[256];
    const int b = blockIdx.x, tid = threadIdx.x;
    const float* row = score + (size_t)b * kS;

    float mx = -3.402823466e38f;
    for (int s = tid; s < kS; s += 256) mx = fmaxf(mx, row[s]);
    red[tid] = mx; __syncthreads();
    for (int off = 128; off; off >>= 1) {
        if (tid < off) red[tid] = fmaxf(red[tid], red[tid + off]);
        __syncthreads();
    }
    mx = red[0]; __syncthreads();

    float sum = 0.f;
    for (int s = tid; s < kS; s += 256) sum += __expf(row[s] - mx);
    red[tid] = sum; __syncthreads();
    for (int off = 128; off; off >>= 1) {
        if (tid < off) red[tid] += red[tid + off];
        __syncthreads();
    }
    const float inv = 1.0f / red[0];

    for (int s = tid; s < kS; s += 256)
        attn[(size_t)b * kS + s] = __expf(row[s] - mx) * inv;
}

// ---------------------------------------------------------------------------
// Kernel 4: context partials (s-split, bandwidth-bound pass over enc, f32)
// ---------------------------------------------------------------------------
__global__ void context_partial_kernel(const float* __restrict__ attn,
                                       const float* __restrict__ enc,
                                       float* __restrict__ partial) {
    const int h = blockIdx.x * blockDim.x + threadIdx.x;
    const int b = blockIdx.y;
    const int z = blockIdx.z;
    const int sBeg = z * (kS / kSChunk);
    const int sEnd = sBeg + (kS / kSChunk);
    const float* a = attn + (size_t)b * kS;
    const float* e = enc + ((size_t)b * kS + sBeg) * kH + h;
    float acc = 0.f;
    #pragma unroll 4
    for (int s = sBeg; s < sEnd; ++s, e += kH) acc += a[s] * (*e);
    partial[((size_t)z * kB + b) * kH + h] = acc;
}

__global__ void context_reduce_kernel(const float* __restrict__ partial,
                                      float* __restrict__ ctx) {
    const int idx = blockIdx.x * blockDim.x + threadIdx.x;   // kB*kH
    float acc = 0.f;
    #pragma unroll
    for (int z = 0; z < kSChunk; ++z)
        acc += partial[(size_t)z * kB * kH + idx];
    ctx[idx] = acc;
}

// ---------------------------------------------------------------------------
extern "C" void kernel_launch(void* const* d_in, const int* in_sizes, int n_in,
                              void* d_out, int out_size, void* d_ws, size_t ws_size,
                              hipStream_t stream) {
    (void)in_sizes; (void)n_in; (void)out_size; (void)ws_size;

    const float* hidden = (const float*)d_in[0];   // [B,H]
    const float* enc    = (const float*)d_in[1];   // [B,S,H]
    const float* W_a    = (const float*)d_in[2];   // [H,H]
    const float* U_a    = (const float*)d_in[3];   // [H,H]
    const float* v_a    = (const float*)d_in[4];   // [H]

    float* ctx  = (float*)d_out;                   // [B,H]
    float* attn = (float*)d_out + kB * kH;         // [B,S]

    // workspace layout (~3.4 MB total)
    float* wh      = (float*)d_ws;                           // kB*kH floats
    float* score   = wh + kB * kH;                           // kB*kS floats
    float* partial = score + (size_t)kB * kS;                // kSChunk*kB*kH floats
    unsigned short* U_bf = (unsigned short*)(partial + (size_t)kSChunk * kB * kH);

    cvt_ua_kernel<<<(kH * kH / 4) / 256, 256, 0, stream>>>(U_a, U_bf);

    wh_kernel<<<(kB * kH) / 256, 256, 0, stream>>>(hidden, W_a, wh);

    dim3 g2(kS / kSTileRows, kB);                  // (16, 32) blocks
    size_t ldsBytes = (size_t)kSTileRows * kAStride * sizeof(unsigned short); // ~258KB
    score_kernel<<<g2, kWaves * 32, ldsBytes, stream>>>(enc, U_bf, wh, v_a, score);

    softmax_kernel<<<kB, 256, 0, stream>>>(score, attn);

    dim3 g4(kH / 256, kB, kSChunk);
    context_partial_kernel<<<g4, 256, 0, stream>>>(attn, enc, partial);
    context_reduce_kernel<<<(kB * kH) / 256, 256, 0, stream>>>(partial, ctx);
}
